// TSAE_49426483642516
// MI455X (gfx1250) — matrix operand order repaired and checked by reading, not compiled, not run on hardware
//
#include <hip/hip_runtime.h>
#include <stdint.h>

typedef unsigned short u16;
typedef __attribute__((ext_vector_type(16))) __bf16 v16bf;
typedef __attribute__((ext_vector_type(8)))  float  v8f;

union FragBF { v16bf v; u16 u[16]; };

__device__ __forceinline__ u16 f2bf(float f) {
    union { float f; unsigned u; } x; x.f = f;
    unsigned r = x.u + 0x7FFFu + ((x.u >> 16) & 1u);
    return (u16)(r >> 16);
}

__device__ __forceinline__ v8f wmma_bf16(v16bf a, v16bf b, v8f c) {
    return __builtin_amdgcn_wmma_f32_16x16x32_bf16(false, a, false, b, (short)0, c, false, false);
}

// Async global->LDS (CDNA5 TDM-adjacent path, ASYNCcnt-tracked). GVS mode:
// mem_addr = SGPR_base + VGPR_offset; per-lane 16B moved straight into LDS.
__device__ __forceinline__ void async_ld_b128(unsigned lds_addr, unsigned goff, const void* sbase) {
    asm volatile("global_load_async_to_lds_b128 %0, %1, %2"
                 :: "v"(lds_addr), "v"(goff), "s"(sbase)
                 : "memory");
}
__device__ __forceinline__ void wait_async0() {
    asm volatile("s_wait_asynccnt 0x0" ::: "memory");
}
__device__ __forceinline__ unsigned lds_addr_of(const void* p) {
    return (unsigned)(uintptr_t)p;  // low 32 bits of generic pointer = LDS byte address
}

// A-fragment (16x32, M x K): lane<16 -> row=lane, K k0..k0+15;
// lane>=16 -> row=lane-16, K k0+16..k0+31. p points at element (m0, k0), row-major.
__device__ __forceinline__ v16bf load_a_frag(const u16* p, int ld) {
    int lane = threadIdx.x & 31;
    int half = lane >> 4, mr = lane & 15;
    const u16* q = p + (size_t)mr * ld + half * 16;
    FragBF f;
#pragma unroll
    for (int s = 0; s < 16; ++s) f.u[s] = q[s];
    return f.v;
}

// B-fragment (32x16, K x N) from row-major [n][k] storage (i.e. element (k,n) = p[n*ld + k]):
// lane<16 -> col n=lane, K k0..k0+15; lane>=16 -> col n=lane-16, K k0+16..k0+31.
__device__ __forceinline__ v16bf load_b_frag(const u16* p, int ld) {
    int lane = threadIdx.x & 31;
    int half = lane >> 4, mr = lane & 15;
    const u16* q = p + (size_t)mr * ld + half * 16;
    FragBF f;
#pragma unroll
    for (int s = 0; s < 16; ++s) f.u[s] = q[s];
    return f.v;
}

// ---------------- elementwise converters ----------------
__global__ void cvt_f32_bf16(const float* __restrict__ src, u16* __restrict__ dst, int n) {
    int i = blockIdx.x * blockDim.x + threadIdx.x;
    if (i < n) dst[i] = f2bf(src[i]);
}

__global__ void make_src(const float* __restrict__ zL, const float* __restrict__ xprior,
                         const float* __restrict__ bias_pre, u16* __restrict__ dst,
                         int total, int H) {
    int i = blockIdx.x * blockDim.x + threadIdx.x;
    if (i < total) {
        int c = i & (H - 1);  // H is a power of two (512)
        dst[i] = f2bf(zL[i] - xprior[i] - bias_pre[c]);
    }
}

// ---------------- build x0 + layernorm (one wave per token) ----------------
__global__ __launch_bounds__(256) void build_x0_ln(
    const float* __restrict__ zL, const float* __restrict__ qtok,
    const float* __restrict__ gamma, const float* __restrict__ beta,
    float* __restrict__ x0, u16* __restrict__ ln,
    int B, int D, int L, int H) {
    int t = blockIdx.x * 8 + (threadIdx.x >> 5);
    int lane = threadIdx.x & 31;
    int DL = D * L;
    int b = t / DL, rem = t - b * DL;
    int d = rem / L, l = rem - d * L;
    const float* src = (d == 0) ? qtok
                                : (zL + ((size_t)((b * D + (d - 1)) * L + l)) * H);
    float vals[16];
    float s = 0.f, ss = 0.f;
    int cnt = 0;
    for (int c = lane; c < H; c += 32) {
        float v = src[c];
        x0[(size_t)t * H + c] = v;
        vals[cnt++] = v;
        s += v; ss += v * v;
    }
    for (int off = 1; off < 32; off <<= 1) { s += __shfl_xor(s, off, 32); ss += __shfl_xor(ss, off, 32); }
    float mu = s / H;
    float var = ss / H - mu * mu;
    float rstd = rsqrtf(var + 1e-5f);
    cnt = 0;
    for (int c = lane; c < H; c += 32) {
        float v = (vals[cnt++] - mu) * rstd * gamma[c] + beta[c];
        ln[(size_t)t * H + c] = f2bf(v);
    }
}

__global__ __launch_bounds__(256) void ln_kernel(
    const float* __restrict__ x, const float* __restrict__ gamma, const float* __restrict__ beta,
    u16* __restrict__ ln, int H) {
    int t = blockIdx.x * 8 + (threadIdx.x >> 5);
    int lane = threadIdx.x & 31;
    float vals[16];
    float s = 0.f, ss = 0.f;
    int cnt = 0;
    for (int c = lane; c < H; c += 32) {
        float v = x[(size_t)t * H + c];
        vals[cnt++] = v;
        s += v; ss += v * v;
    }
    for (int off = 1; off < 32; off <<= 1) { s += __shfl_xor(s, off, 32); ss += __shfl_xor(ss, off, 32); }
    float mu = s / H;
    float var = ss / H - mu * mu;
    float rstd = rsqrtf(var + 1e-5f);
    cnt = 0;
    for (int c = lane; c < H; c += 32) {
        float v = (vals[cnt++] - mu) * rstd * gamma[c] + beta[c];
        ln[(size_t)t * H + c] = f2bf(v);
    }
}

// ---- WMMA GEMM: C[M,N] = A[M,K] @ W[N,K]^T, 128x128 tile, KTILE=64, async double-buffered ----
// MODE 0: bf16 out. MODE 1: f32 out + residual. MODE 2: f32 out + bias + relu.
template <int MODE>
__global__ __launch_bounds__(256) void gemm_kernel(
    const u16* __restrict__ A, const u16* __restrict__ W,
    u16* __restrict__ outb, float* __restrict__ outf,
    const float* __restrict__ res, const float* __restrict__ bias,
    int M, int Nn, int K) {
    __shared__ u16 As[2][128 * 64];
    __shared__ u16 Bs[2][128 * 64];
    int tid = threadIdx.x;
    int lane = tid & 31, wid = tid >> 5;
    int half = lane >> 4, mr = lane & 15;
    int wm = wid & 3, wn = wid >> 2;
    int m0b = blockIdx.y * 128, n0b = blockIdx.x * 128;
    v8f acc[2][4] = {};

    // each thread stages 4 x 16B chunks per operand per tile (128 rows x 128B)
    auto stage = [&](int kt, int buf) {
#pragma unroll
        for (int c = 0; c < 4; ++c) {
            int ci = tid * 4 + c;            // 0..1023
            int row = ci >> 3, kc = (ci & 7) * 8;
            unsigned ga = (unsigned)(((m0b + row) * K + kt + kc) * 2);
            async_ld_b128(lds_addr_of(&As[buf][row * 64 + kc]), ga, A);
            unsigned gb = (unsigned)(((n0b + row) * K + kt + kc) * 2);
            async_ld_b128(lds_addr_of(&Bs[buf][row * 64 + kc]), gb, W);
        }
    };

    stage(0, 0);
    wait_async0();
    __syncthreads();
    int buf = 0;
    for (int kt = 0; kt < K; kt += 64) {
        if (kt + 64 < K) stage(kt + 64, buf ^ 1);  // prefetch next tile while computing
#pragma unroll
        for (int ks = 0; ks < 2; ++ks) {
            v16bf a0 = load_a_frag(As[buf] + (wm * 32 + 0) * 64 + ks * 32, 64);
            v16bf a1 = load_a_frag(As[buf] + (wm * 32 + 16) * 64 + ks * 32, 64);
#pragma unroll
            for (int nf = 0; nf < 4; ++nf) {
                v16bf bb = load_b_frag(Bs[buf] + (wn * 64 + nf * 16) * 64 + ks * 32, 64);
                acc[0][nf] = wmma_bf16(a0, bb, acc[0][nf]);
                acc[1][nf] = wmma_bf16(a1, bb, acc[1][nf]);
            }
        }
        wait_async0();
        __syncthreads();
        buf ^= 1;
    }
#pragma unroll
    for (int mf = 0; mf < 2; ++mf)
#pragma unroll
        for (int nf = 0; nf < 4; ++nf)
#pragma unroll
            for (int r = 0; r < 8; ++r) {
                int row = m0b + wm * 32 + mf * 16 + half * 8 + r;
                int col = n0b + wn * 64 + nf * 16 + mr;
                size_t o = (size_t)row * Nn + col;
                float v = acc[mf][nf][r];
                if constexpr (MODE == 0) {
                    outb[o] = f2bf(v);
                } else if constexpr (MODE == 1) {
                    outf[o] = v + res[o];
                } else {
                    float t2 = v + bias[col];
                    outf[o] = t2 > 0.f ? t2 : 0.f;
                }
            }
}

// ---------------- layer attention (non-causal, S=128, hd=64), block = (b*D+d, head) -------
__global__ __launch_bounds__(256) void attn_layer(
    const u16* __restrict__ Q, const u16* __restrict__ Kb, const u16* __restrict__ Vb,
    u16* __restrict__ Ob, int H) {
    __shared__ u16 Ks[128 * 64];   // [key][dim]
    __shared__ u16 VTs[64 * 128];  // [dim][key]
    __shared__ u16 Pw[8][16 * 128];
    int head = blockIdx.x & 7;
    int bd = blockIdx.x >> 3;
    int tid = threadIdx.x, lane = tid & 31, wid = tid >> 5;
    int half = lane >> 4, mr = lane & 15;
    size_t base = (size_t)bd * 128;
    int hc = head * 64;
    // K tile via async global->LDS (contiguous 16B chunks)
#pragma unroll
    for (int c = 0; c < 4; ++c) {
        int ci = tid * 4 + c;               // 0..1023
        int n = ci >> 3, kc = (ci & 7) * 8;
        unsigned goff = (unsigned)((((unsigned)(base + n)) * H + hc + kc) * 2);
        async_ld_b128(lds_addr_of(&Ks[n * 64 + kc]), goff, Kb);
    }
    // V transposed (scatter) via normal path
    for (int i = tid; i < 128 * 64; i += 256) {
        int n = i >> 6, k = i & 63;
        VTs[k * 128 + n] = Vb[(base + n) * H + hc + k];
    }
    wait_async0();
    __syncthreads();
    int m0 = wid * 16;
    const float scale = 0.125f;  // 1/sqrt(64)
    v16bf aq0 = load_a_frag(Q + (base + m0) * H + hc + 0, H);
    v16bf aq1 = load_a_frag(Q + (base + m0) * H + hc + 32, H);
    v8f s[8] = {};
#pragma unroll
    for (int nf = 0; nf < 8; ++nf) {
        v16bf b0 = load_b_frag(Ks + (nf * 16) * 64 + 0, 64);
        v16bf b1 = load_b_frag(Ks + (nf * 16) * 64 + 32, 64);
        s[nf] = wmma_bf16(aq0, b0, s[nf]);
        s[nf] = wmma_bf16(aq1, b1, s[nf]);
    }
    float rs[8];
    u16* Pme = Pw[wid];
#pragma unroll
    for (int r = 0; r < 8; ++r) {
        float m = -1e30f;
#pragma unroll
        for (int nf = 0; nf < 8; ++nf) m = fmaxf(m, s[nf][r]);
        for (int off = 1; off < 16; off <<= 1) m = fmaxf(m, __shfl_xor(m, off, 32));
        float sum = 0.f;
        int rowin = half * 8 + r;
#pragma unroll
        for (int nf = 0; nf < 8; ++nf) {
            float e = __expf((s[nf][r] - m) * scale);
            sum += e;
            Pme[rowin * 128 + nf * 16 + mr] = f2bf(e);
        }
        for (int off = 1; off < 16; off <<= 1) sum += __shfl_xor(sum, off, 32);
        rs[r] = sum;
    }
    __syncthreads();
    v8f o[4] = {};
#pragma unroll
    for (int kst = 0; kst < 4; ++kst) {
        v16bf ap = load_a_frag(Pme + kst * 32, 128);
#pragma unroll
        for (int nf = 0; nf < 4; ++nf) {
            v16bf bv = load_b_frag(VTs + (nf * 16) * 128 + kst * 32, 128);
            o[nf] = wmma_bf16(ap, bv, o[nf]);
        }
    }
#pragma unroll
    for (int nf = 0; nf < 4; ++nf)
#pragma unroll
        for (int r = 0; r < 8; ++r) {
            size_t rowtok = base + m0 + half * 8 + r;
            int col = hc + nf * 16 + mr;
            Ob[rowtok * H + col] = f2bf(o[nf][r] / rs[r]);
        }
}

// ---------------- depth attention (causal, S=D=16, hd=64), block = (b,l), wave = head ------
__global__ __launch_bounds__(256) void attn_depth(
    const u16* __restrict__ Q, const u16* __restrict__ Kb, const u16* __restrict__ Vb,
    u16* __restrict__ Ob, int B, int D, int L, int H) {
    __shared__ u16 VT[8][64][16];   // [head][dim][key]
    __shared__ u16 P[8][16 * 32];   // zero-padded K to 32
    int bl = blockIdx.x;
    int b = bl / L, l = bl - b * L;
    int tid = threadIdx.x, lane = tid & 31, wid = tid >> 5;  // wid = head
    int half = lane >> 4, mr = lane & 15;
    int strideD = L * H;
    size_t base0 = ((size_t)b * D * L + l) * H;
    int hc = wid * 64;
    for (int i = lane; i < 16 * 64; i += 32) {
        int d = i >> 6, dv = i & 63;
        VT[wid][dv][d] = Vb[base0 + (size_t)d * strideD + hc + dv];
    }
    __syncthreads();
    v16bf a0 = load_a_frag(Q + base0 + hc + 0, strideD);
    v16bf a1 = load_a_frag(Q + base0 + hc + 32, strideD);
    v16bf k0 = load_b_frag(Kb + base0 + hc + 0, strideD);
    v16bf k1 = load_b_frag(Kb + base0 + hc + 32, strideD);
    v8f sc = {};
    sc = wmma_bf16(a0, k0, sc);
    sc = wmma_bf16(a1, k1, sc);
    const float scale = 0.125f;
    float rs[8];
    u16* Pme = P[wid];
#pragma unroll
    for (int r = 0; r < 8; ++r) {
        int mrow = half * 8 + r;
        float v = (mr <= mrow) ? sc[r] : -1e30f;
        float m = v;
        for (int off = 1; off < 16; off <<= 1) m = fmaxf(m, __shfl_xor(m, off, 32));
        float e = __expf((v - m) * scale);
        float sum = e;
        for (int off = 1; off < 16; off <<= 1) sum += __shfl_xor(sum, off, 32);
        rs[r] = sum;
        Pme[mrow * 32 + mr] = f2bf(e);
        Pme[mrow * 32 + 16 + mr] = 0;
    }
    __syncthreads();
    v16bf ap = load_a_frag(Pme, 32);
    v8f o[4] = {};
#pragma unroll
    for (int nf = 0; nf < 4; ++nf) {
        FragBF fb;
        if (half == 0) {
            const u16* q = &VT[wid][nf * 16 + mr][0];
#pragma unroll
            for (int s2 = 0; s2 < 16; ++s2) fb.u[s2] = q[s2];
        } else {
#pragma unroll
            for (int s2 = 0; s2 < 16; ++s2) fb.u[s2] = 0;
        }
        o[nf] = wmma_bf16(ap, fb.v, o[nf]);
    }
#pragma unroll
    for (int nf = 0; nf < 4; ++nf)
#pragma unroll
        for (int r = 0; r < 8; ++r) {
            int mrow = half * 8 + r;
            Ob[base0 + (size_t)mrow * strideD + hc + nf * 16 + mr] = f2bf(o[nf][r] / rs[r]);
        }
}

// ---------------- top-k (k=64) per row via 4-pass radix select on non-negative floats ------
__global__ __launch_bounds__(256) void topk_kernel(float* __restrict__ z, int F, int k0) {
    __shared__ float row[4096];
    __shared__ unsigned hist[256];
    __shared__ unsigned sh_pref;
    __shared__ int sh_k;
    size_t base = (size_t)blockIdx.x * F;
    int tid = threadIdx.x;
    for (int i = tid; i < F; i += 256) row[i] = z[base + i];
    if (tid == 0) { sh_pref = 0; sh_k = k0; }
    __syncthreads();
    for (int pass = 0; pass < 4; ++pass) {
        int shift = 24 - pass * 8;
        hist[tid] = 0;
        __syncthreads();
        unsigned pref = sh_pref;
        for (int i = tid; i < F; i += 256) {
            unsigned u = __float_as_uint(row[i]);
            bool cand = (pass == 0) || ((u >> (shift + 8)) == pref);
            if (cand) atomicAdd(&hist[(u >> shift) & 255u], 1u);
        }
        __syncthreads();
        if (tid == 0) {
            int k = sh_k, acc = 0, sel = 0;
            for (int bb = 255; bb >= 0; --bb) {
                int c = (int)hist[bb];
                if (acc + c >= k) { sel = bb; sh_k = k - acc; break; }
                acc += c;
            }
            sh_pref = (sh_pref << 8) | (unsigned)sel;
        }
        __syncthreads();
    }
    unsigned thr = sh_pref;
    for (int i = tid; i < F; i += 256) {
        unsigned u = __float_as_uint(row[i]);
        z[base + i] = (u >= thr) ? row[i] : 0.f;
    }
}

// ------------------------------------ launcher ---------------------------------------------
extern "C" void kernel_launch(void* const* d_in, const int* in_sizes, int n_in,
                              void* d_out, int out_size, void* d_ws, size_t ws_size,
                              hipStream_t stream) {
    (void)in_sizes; (void)n_in; (void)out_size; (void)ws_size;
    const int B = 8, D = 16, L = 128, H = 512, F = 4096, NH = 8;
    const int Ntok = B * D * L;  // 16384

    const float* zL      = (const float*)d_in[0];
    const float* wq_l    = (const float*)d_in[1];
    const float* wk_l    = (const float*)d_in[2];
    const float* wv_l    = (const float*)d_in[3];
    const float* wo_l    = (const float*)d_in[4];
    const float* gamma_l = (const float*)d_in[5];
    const float* beta_l  = (const float*)d_in[6];
    const float* wq_d    = (const float*)d_in[7];
    const float* wk_d    = (const float*)d_in[8];
    const float* wv_d    = (const float*)d_in[9];
    const float* wo_d    = (const float*)d_in[10];
    const float* gamma_d = (const float*)d_in[11];
    const float* beta_d  = (const float*)d_in[12];
    const float* dict_enc = (const float*)d_in[13];
    const float* bias_pre = (const float*)d_in[14];
    const float* bias_enc = (const float*)d_in[15];
    const float* qtok     = (const float*)d_in[16];

    float* z_out = (float*)d_out;
    float* xprior_out = z_out + (size_t)Ntok * F;

    char* wsp = (char*)d_ws;
    auto alloc = [&](size_t bytes) -> char* {
        char* p = wsp;
        wsp += (bytes + 255) & ~(size_t)255;
        return p;
    };
    u16* wqlb = (u16*)alloc((size_t)H * H * 2);
    u16* wklb = (u16*)alloc((size_t)H * H * 2);
    u16* wvlb = (u16*)alloc((size_t)H * H * 2);
    u16* wolb = (u16*)alloc((size_t)H * H * 2);
    u16* wqdb = (u16*)alloc((size_t)H * H * 2);
    u16* wkdb = (u16*)alloc((size_t)H * H * 2);
    u16* wvdb = (u16*)alloc((size_t)H * H * 2);
    u16* wodb = (u16*)alloc((size_t)H * H * 2);
    u16* dictb = (u16*)alloc((size_t)F * H * 2);
    float* x0 = (float*)alloc((size_t)Ntok * H * 4);
    u16* ln = (u16*)alloc((size_t)Ntok * H * 2);
    u16* qb = (u16*)alloc((size_t)Ntok * H * 2);
    u16* kb = (u16*)alloc((size_t)Ntok * H * 2);
    u16* vb = (u16*)alloc((size_t)Ntok * H * 2);
    u16* ab = (u16*)alloc((size_t)Ntok * H * 2);
    float* x1 = (float*)alloc((size_t)Ntok * H * 4);

    // weights -> bf16
    int wn = H * H;
    int wg = (wn + 255) / 256;
    cvt_f32_bf16<<<wg, 256, 0, stream>>>(wq_l, wqlb, wn);
    cvt_f32_bf16<<<wg, 256, 0, stream>>>(wk_l, wklb, wn);
    cvt_f32_bf16<<<wg, 256, 0, stream>>>(wv_l, wvlb, wn);
    cvt_f32_bf16<<<wg, 256, 0, stream>>>(wo_l, wolb, wn);
    cvt_f32_bf16<<<wg, 256, 0, stream>>>(wq_d, wqdb, wn);
    cvt_f32_bf16<<<wg, 256, 0, stream>>>(wk_d, wkdb, wn);
    cvt_f32_bf16<<<wg, 256, 0, stream>>>(wv_d, wvdb, wn);
    cvt_f32_bf16<<<wg, 256, 0, stream>>>(wo_d, wodb, wn);
    int dn = F * H;
    cvt_f32_bf16<<<(dn + 255) / 256, 256, 0, stream>>>(dict_enc, dictb, dn);

    // x0 + LN
    build_x0_ln<<<Ntok / 8, 256, 0, stream>>>(zL, qtok, gamma_l, beta_l, x0, ln, B, D, L, H);

    dim3 gP(H / 128, Ntok / 128);  // (4, 128)
    gemm_kernel<0><<<gP, 256, 0, stream>>>(ln, wqlb, qb, nullptr, nullptr, nullptr, Ntok, H, H);
    gemm_kernel<0><<<gP, 256, 0, stream>>>(ln, wklb, kb, nullptr, nullptr, nullptr, Ntok, H, H);
    gemm_kernel<0><<<gP, 256, 0, stream>>>(ln, wvlb, vb, nullptr, nullptr, nullptr, Ntok, H, H);

    attn_layer<<<B * D * NH, 256, 0, stream>>>(qb, kb, vb, ab, H);

    gemm_kernel<1><<<gP, 256, 0, stream>>>(ab, wolb, nullptr, x1, x0, nullptr, Ntok, H, H);

    ln_kernel<<<Ntok / 8, 256, 0, stream>>>(x1, gamma_d, beta_d, ln, H);

    gemm_kernel<0><<<gP, 256, 0, stream>>>(ln, wqdb, qb, nullptr, nullptr, nullptr, Ntok, H, H);
    gemm_kernel<0><<<gP, 256, 0, stream>>>(ln, wkdb, kb, nullptr, nullptr, nullptr, Ntok, H, H);
    gemm_kernel<0><<<gP, 256, 0, stream>>>(ln, wvdb, vb, nullptr, nullptr, nullptr, Ntok, H, H);

    attn_depth<<<B * L, 256, 0, stream>>>(qb, kb, vb, ab, B, D, L, H);

    gemm_kernel<1><<<gP, 256, 0, stream>>>(ab, wodb, nullptr, xprior_out, x1, nullptr, Ntok, H, H);

    // encoder
    int tot = Ntok * H;
    make_src<<<(tot + 255) / 256, 256, 0, stream>>>(zL, xprior_out, bias_pre, ln, tot, H);
    dim3 gE(F / 128, Ntok / 128);  // (32, 128)
    gemm_kernel<2><<<gE, 256, 0, stream>>>(ln, dictb, nullptr, z_out, nullptr, bias_enc, Ntok, F, H);

    topk_kernel<<<Ntok, 256, 0, stream>>>(z_out, F, 64);
}
